// FRCell_36455682408727
// MI455X (gfx1250) — compile-verified, matrix operand order checked by reading, not run on hardware
//
#include <hip/hip_runtime.h>

// ---------------------------------------------------------------------------
// FRCell for MI455X (gfx1250): wave32 + v_wmma_f32_16x16x32_bf16 for the MLPs.
// ---------------------------------------------------------------------------

typedef __bf16 bf16;
typedef __attribute__((ext_vector_type(16))) __bf16 v16bf;
typedef __attribute__((ext_vector_type(8)))  float  v8f;

__device__ __forceinline__ bf16 tobf(float f) { return (bf16)f; }

#define WAVES_PB 8   // 256 threads = 8 wave32

// ---------------------------------------------------------------------------
// Stage 1: per-graph node counts
__global__ void frc_count_kernel(const long long* __restrict__ batch, float* __restrict__ cnt, int n) {
    int i = blockIdx.x * blockDim.x + threadIdx.x;
    if (i < n) atomicAdd(&cnt[batch[i]], 1.0f);
}

__global__ void frc_kpg_kernel(const float* __restrict__ cnt, float* __restrict__ kpg, int b) {
    int i = blockIdx.x * blockDim.x + threadIdx.x;
    if (i < b) kpg[i] = rsqrtf(fmaxf(cnt[i], 1.0f));
}

// ---------------------------------------------------------------------------
// Stage 2: per-edge attraction force + out-degree (f32 global atomics, L2-resident)
__global__ void frc_fattr_kernel(const float* __restrict__ h,
                                 const long long* __restrict__ row,
                                 const long long* __restrict__ col,
                                 const long long* __restrict__ batch,
                                 const float* __restrict__ kpg,
                                 float* __restrict__ F, float* __restrict__ deg, int E_) {
    int e = blockIdx.x * blockDim.x + threadIdx.x;
    if (e >= E_) return;
    long long r = row[e], c = col[e];
    float dx = h[r * 64 + 0] - h[c * 64 + 0];
    float dy = h[r * 64 + 1] - h[c * 64 + 1];
    float dist = sqrtf(dx * dx + dy * dy + 1e-12f) + 0.01f;   // + EPS
    float s = -(dist / kpg[batch[r]]);
    float ax = s * dx, ay = s * dy;
    atomicAdd(&F[r * 2 + 0],  ax);
    atomicAdd(&F[r * 2 + 1],  ay);
    atomicAdd(&F[c * 2 + 0], -ax);
    atomicAdd(&F[c * 2 + 1], -ay);
    atomicAdd(&deg[r], 1.0f);
}

// ---------------------------------------------------------------------------
// Stage 3: per-graph all-pairs repulsion (coords staged in LDS; 1 block/graph)
__global__ __launch_bounds__(512) void frc_frep_kernel(const float* __restrict__ h,
                                                       const float* __restrict__ kpg,
                                                       float* __restrict__ F, int NG_) {
    __shared__ float sx[512];
    __shared__ float sy[512];
    int b = blockIdx.x;
    int i = threadIdx.x;
    long long node = (long long)b * NG_ + i;
    if (i < NG_) { sx[i] = h[node * 64 + 0]; sy[i] = h[node * 64 + 1]; }
    __syncthreads();
    if (i >= NG_) return;
    float k2 = kpg[b]; k2 = k2 * k2;
    float xi = sx[i], yi = sy[i];
    float axx = 0.f, ayy = 0.f;
    for (int j = 0; j < NG_; ++j) {
        if (j == i) continue;
        float dx = xi - sx[j], dy = yi - sy[j];
        float d = sqrtf(dx * dx + dy * dy) + 0.01f;           // sqrt(sq)+EPS off-diagonal
        float m = k2 / (d * d);                               // mag/distg
        axx += m * dx; ayy += m * dy;
    }
    // fattr kernel finished earlier on the stream -> plain RMW is safe (unique node per thread)
    F[node * 2 + 0] += axx;
    F[node * 2 + 1] += ayy;
}

// ---------------------------------------------------------------------------
// Stage 4: edge message MLP  m = relu(relu(h[col]@W1+b1)@W2+b2), agg[row]+=m
// One wave per 16-edge tile; bf16 WMMA with f32 accumulate.
// A-fragment layout per ISA "16-bit A-Matrix 16x32":
//   lane L<16 : row M=L,   elems j<8 -> K=j,      j>=8 -> K=16+(j-8)
//   lane L>=16: row M=L-16,elems j<8 -> K=8+j,    j>=8 -> K=24+(j-8)
// C/D layout: vgpr r -> row = r + 8*(lane>=16), col = lane&15.
__global__ __launch_bounds__(256) void frc_edge_mlp_kernel(
        const float* __restrict__ h,
        const long long* __restrict__ row, const long long* __restrict__ col,
        const float* __restrict__ W1, const float* __restrict__ b1,
        const float* __restrict__ W2, const float* __restrict__ b2,
        float* __restrict__ agg, int E_, int nTiles) {
    __shared__ bf16 sW1[64 * 64];
    __shared__ bf16 sW2[64 * 64];
    __shared__ float sb1[64], sb2[64];
    __shared__ bf16 sY[WAVES_PB][16 * 64];     // per-wave D->A re-stripe buffer

    int tid = threadIdx.x;
    for (int i = tid; i < 64 * 64; i += blockDim.x) { sW1[i] = tobf(W1[i]); sW2[i] = tobf(W2[i]); }
    for (int i = tid; i < 64; i += blockDim.x)      { sb1[i] = b1[i]; sb2[i] = b2[i]; }
    __syncthreads();

    int wave = tid >> 5, lane = tid & 31;
    int m0 = lane & 15, khalf = lane >> 4;
    bf16* myY = sY[wave];

    for (int t = blockIdx.x * WAVES_PB + wave; t < nTiles; t += (int)gridDim.x * WAVES_PB) {
        // ---- gather X = h[col] tile (16 edges x 64 feats) as two bf16 A-fragments
        long long eA = (long long)t * 16 + m0;
        long long cn = (eA < E_) ? col[eA] : 0;
        const float* xr = h + cn * 64;
        v16bf A0, A1;
#pragma unroll
        for (int j = 0; j < 16; ++j) {
            int k = ((j < 8) ? 0 : 16) + khalf * 8 + (j & 7);
            A0[j] = tobf(xr[k]);
            A1[j] = tobf(xr[32 + k]);
        }
        // ---- layer 1: Y = relu(X@W1 + b1), written to LDS as bf16 (A layout for layer 2)
#pragma unroll
        for (int nt = 0; nt < 4; ++nt) {
            v8f acc;
            float bv = sb1[nt * 16 + m0];
#pragma unroll
            for (int j = 0; j < 8; ++j) acc[j] = bv;
            v16bf B0, B1;
#pragma unroll
            for (int j = 0; j < 16; ++j) {
                int kr = ((j < 8) ? 0 : 16) + khalf * 8 + (j & 7);
                B0[j] = sW1[kr * 64 + nt * 16 + m0];
                B1[j] = sW1[(kr + 32) * 64 + nt * 16 + m0];
            }
            acc = __builtin_amdgcn_wmma_f32_16x16x32_bf16(false, A0, false, B0, (short)0, acc, false, false);
            acc = __builtin_amdgcn_wmma_f32_16x16x32_bf16(false, A1, false, B1, (short)0, acc, false, false);
#pragma unroll
            for (int j = 0; j < 8; ++j) {
                float v = acc[j] > 0.f ? acc[j] : 0.f;
                myY[(j + 8 * khalf) * 64 + nt * 16 + m0] = tobf(v);   // D layout -> row-major tile
            }
        }
        // same-wave DS ops are in-order: safe to reload without a barrier
        v16bf C0, C1;
#pragma unroll
        for (int j = 0; j < 16; ++j) {
            int k = ((j < 8) ? 0 : 16) + khalf * 8 + (j & 7);
            C0[j] = myY[m0 * 64 + k];
            C1[j] = myY[m0 * 64 + 32 + k];
        }
        // ---- layer 2: m = relu(Y@W2 + b2), scatter into agg[row]
#pragma unroll
        for (int nt = 0; nt < 4; ++nt) {
            v8f acc;
            float bv = sb2[nt * 16 + m0];
#pragma unroll
            for (int j = 0; j < 8; ++j) acc[j] = bv;
            v16bf B0, B1;
#pragma unroll
            for (int j = 0; j < 16; ++j) {
                int kr = ((j < 8) ? 0 : 16) + khalf * 8 + (j & 7);
                B0[j] = sW2[kr * 64 + nt * 16 + m0];
                B1[j] = sW2[(kr + 32) * 64 + nt * 16 + m0];
            }
            acc = __builtin_amdgcn_wmma_f32_16x16x32_bf16(false, C0, false, B0, (short)0, acc, false, false);
            acc = __builtin_amdgcn_wmma_f32_16x16x32_bf16(false, C1, false, B1, (short)0, acc, false, false);
#pragma unroll
            for (int j = 0; j < 8; ++j) {
                float v = acc[j] > 0.f ? acc[j] : 0.f;
                long long eS = (long long)t * 16 + (j + 8 * khalf);
                if (eS < E_) {
                    long long rn = row[eS];
                    atomicAdd(&agg[rn * 64 + nt * 16 + m0], v);
                }
            }
        }
    }
}

// ---------------------------------------------------------------------------
// Stage 5: agg /= max(deg, 1)
__global__ void frc_norm_agg_kernel(const float* __restrict__ deg, float* __restrict__ agg, long long n) {
    long long i = (long long)blockIdx.x * blockDim.x + threadIdx.x;
    if (i >= n) return;
    agg[i] /= fmaxf(deg[i >> 6], 1.0f);
}

// ---------------------------------------------------------------------------
// Stage 6: node MLP  res = relu([h, agg]@U1 + c1) @ U2 + c2   (N x 2)
__global__ __launch_bounds__(256) void frc_node_mlp_kernel(
        const float* __restrict__ h, const float* __restrict__ agg,
        const float* __restrict__ U1, const float* __restrict__ c1,
        const float* __restrict__ U2, const float* __restrict__ c2,
        float* __restrict__ res, int nTiles) {
    __shared__ bf16 sU1[128 * 64];
    __shared__ float sc1[64];
    __shared__ float sU2[64 * 2];
    __shared__ float sc2[2];
    __shared__ float sYf[WAVES_PB][16 * 64];

    int tid = threadIdx.x;
    for (int i = tid; i < 128 * 64; i += blockDim.x) sU1[i] = tobf(U1[i]);
    for (int i = tid; i < 64; i += blockDim.x)       sc1[i] = c1[i];
    for (int i = tid; i < 128; i += blockDim.x)      sU2[i] = U2[i];
    if (tid < 2) sc2[tid] = c2[tid];
    __syncthreads();

    int wave = tid >> 5, lane = tid & 31;
    int m0 = lane & 15, khalf = lane >> 4;
    float* myY = sYf[wave];

    for (int t = blockIdx.x * WAVES_PB + wave; t < nTiles; t += (int)gridDim.x * WAVES_PB) {
        long long node = (long long)t * 16 + m0;
        const float* xr = h + node * 64;
        const float* ar = agg + node * 64;
        v16bf A[4];                                  // [h | agg] 16x128 -> 4 K-tiles
#pragma unroll
        for (int kt = 0; kt < 4; ++kt) {
            const float* src = (kt < 2) ? (xr + kt * 32) : (ar + (kt - 2) * 32);
#pragma unroll
            for (int j = 0; j < 16; ++j) {
                int k = ((j < 8) ? 0 : 16) + khalf * 8 + (j & 7);
                A[kt][j] = tobf(src[k]);
            }
        }
#pragma unroll
        for (int nt = 0; nt < 4; ++nt) {
            v8f acc;
            float bv = sc1[nt * 16 + m0];
#pragma unroll
            for (int j = 0; j < 8; ++j) acc[j] = bv;
#pragma unroll
            for (int kt = 0; kt < 4; ++kt) {
                v16bf Bf;
#pragma unroll
                for (int j = 0; j < 16; ++j) {
                    int kr = kt * 32 + ((j < 8) ? 0 : 16) + khalf * 8 + (j & 7);
                    Bf[j] = sU1[kr * 64 + nt * 16 + m0];
                }
                acc = __builtin_amdgcn_wmma_f32_16x16x32_bf16(false, A[kt], false, Bf, (short)0, acc, false, false);
            }
#pragma unroll
            for (int j = 0; j < 8; ++j) {
                float v = acc[j] > 0.f ? acc[j] : 0.f;
                myY[(j + 8 * khalf) * 64 + nt * 16 + m0] = v;
            }
        }
        // tiny layer 2 (64 -> 2): lane handles (node m0, output khalf)
        float s = sc2[khalf];
#pragma unroll 8
        for (int k = 0; k < 64; ++k) s += myY[m0 * 64 + k] * sU2[k * 2 + khalf];
        res[((long long)t * 16 + m0) * 2 + khalf] = s;
    }
}

// ---------------------------------------------------------------------------
// Stage 7: delta = alpha*F + res, step-norm clamp, assemble output
__global__ void frc_final_kernel(const float* __restrict__ h, const float* __restrict__ F,
                                 const float* __restrict__ res, const float* __restrict__ alpha,
                                 float* __restrict__ out, int n) {
    int i = blockIdx.x * blockDim.x + threadIdx.x;
    if (i >= n) return;
    float a = alpha[0];
    float dx = a * F[i * 2 + 0] + res[i * 2 + 0];
    float dy = a * F[i * 2 + 1] + res[i * 2 + 1];
    float sn = sqrtf(dx * dx + dy * dy);
    float scale = fminf(0.1f / (sn + 1e-9f), 1.0f);   // CLAMP_STEP
    const float* hr = h + (long long)i * 64;
    float* orow = out + (long long)i * 64;
    orow[0] = hr[0] + dx * scale;
    orow[1] = hr[1] + dy * scale;
#pragma unroll
    for (int k = 2; k < 64; ++k) orow[k] = hr[k];
}

// ---------------------------------------------------------------------------
extern "C" void kernel_launch(void* const* d_in, const int* in_sizes, int n_in,
                              void* d_out, int out_size, void* d_ws, size_t ws_size,
                              hipStream_t stream) {
    const float*     h     = (const float*)d_in[0];
    const long long* ei    = (const long long*)d_in[1];
    const long long* batch = (const long long*)d_in[2];
    const float*     alpha = (const float*)d_in[3];
    const float*     W1    = (const float*)d_in[4];
    const float*     b1    = (const float*)d_in[5];
    const float*     W2    = (const float*)d_in[6];
    const float*     b2    = (const float*)d_in[7];
    const float*     U1    = (const float*)d_in[8];
    const float*     c1    = (const float*)d_in[9];
    const float*     U2    = (const float*)d_in[10];
    const float*     c2    = (const float*)d_in[11];

    const int N_ = in_sizes[0] / 64;
    const int E_ = in_sizes[1] / 2;
    const int B_ = 64;
    const int NG_ = N_ / B_;

    const long long* row = ei;
    const long long* col = ei + E_;

    // workspace layout (floats): F[2N] deg[N] cnt[B] kpg[B] res[2N] agg[64N]
    float* F   = (float*)d_ws;
    float* deg = F + (size_t)2 * N_;
    float* cnt = deg + N_;
    float* kpg = cnt + B_;
    float* res = kpg + B_;
    float* agg = res + (size_t)2 * N_;
    size_t needF = (size_t)69 * N_ + 2 * B_;
    hipMemsetAsync(d_ws, 0, needF * sizeof(float), stream);

    frc_count_kernel<<<(N_ + 255) / 256, 256, 0, stream>>>(batch, cnt, N_);
    frc_kpg_kernel<<<1, 64, 0, stream>>>(cnt, kpg, B_);
    frc_fattr_kernel<<<(E_ + 255) / 256, 256, 0, stream>>>(h, row, col, batch, kpg, F, deg, E_);
    frc_frep_kernel<<<B_, 512, 0, stream>>>(h, kpg, F, NG_);

    int nTilesE = (E_ + 15) / 16;
    int gE = (nTilesE + WAVES_PB - 1) / WAVES_PB;
    frc_edge_mlp_kernel<<<gE, 256, 0, stream>>>(h, row, col, W1, b1, W2, b2, agg, E_, nTilesE);

    long long nAgg = (long long)N_ * 64;
    frc_norm_agg_kernel<<<(int)((nAgg + 255) / 256), 256, 0, stream>>>(deg, agg, nAgg);

    int nTilesN = (N_ + 15) / 16;
    int gN = (nTilesN + WAVES_PB - 1) / WAVES_PB;
    frc_node_mlp_kernel<<<gN, 256, 0, stream>>>(h, agg, U1, c1, U2, c2, res, nTilesN);

    frc_final_kernel<<<(N_ + 255) / 256, 256, 0, stream>>>(h, F, res, alpha, (float*)d_out, N_);
}
